// MultiheadAxialAttention_82678120448795
// MI455X (gfx1250) — compile-verified, matrix operand order checked by reading, not run on hardware
//
#include <hip/hip_runtime.h>
#include <hip/hip_bf16.h>

typedef __bf16 bf16_t;
typedef __attribute__((ext_vector_type(8)))  __bf16 bf16x8;
typedef __attribute__((ext_vector_type(16))) __bf16 v16bf;
typedef __attribute__((ext_vector_type(8)))  float  v8f;

// ---------------------------------------------------------------------------
// WMMA helpers (CDNA5 wave32, v_wmma_f32_16x16x32_bf16)
// ---------------------------------------------------------------------------
__device__ __forceinline__ v8f wmma_bf16(v16bf a, v16bf b, v8f c) {
  // (neg_a, A, neg_b, B, c_mod, C, reuse_a, reuse_b)
  return __builtin_amdgcn_wmma_f32_16x16x32_bf16(false, a, false, b, (short)0, c,
                                                 false, false);
}

// Per-lane 16-bit A/B fragment from a row-major (row, K-contiguous) buffer.
// ISA layout (16-bit 16x32): lanes 0-15 -> K {0..7, 16..23}, lanes 16-31 ->
// K {8..15, 24..31}; row (M or N) = lane % 16.
__device__ __forceinline__ v16bf frag_ld(const bf16_t* rowptr, int half) {
  bf16x8 lo = *(const bf16x8*)(rowptr + half * 8);
  bf16x8 hi = *(const bf16x8*)(rowptr + 16 + half * 8);
  v16bf a;
#pragma unroll
  for (int t = 0; t < 8; ++t) { a[t] = lo[t]; a[8 + t] = hi[t]; }
  return a;
}

// ---------------------------------------------------------------------------
// Tensor Data Mover staging of a 2D bf16 tile into LDS, with hardware row
// padding 16 DWORDs data + 4 DWORDs pad -> 80B LDS row stride (matches the
// [..][40] bf16 LDS arrays used by the WMMA fragment loader).
// D# packing per CDNA5 ISA.
// ---------------------------------------------------------------------------
#if __has_builtin(__builtin_amdgcn_tensor_load_to_lds)
#define HAVE_TDM 1
typedef unsigned int u32x4 __attribute__((ext_vector_type(4)));
typedef int i32x8 __attribute__((ext_vector_type(8)));
typedef int i32x4 __attribute__((ext_vector_type(4)));

__device__ __forceinline__ void tdm_load_tile_bf16(
    const void* gaddr, unsigned int lds_byte_off, int tensor_w, int tensor_h,
    int row_stride_elems, int tile_w, int tile_h) {
  unsigned long long ga = (unsigned long long)(uintptr_t)gaddr;
  u32x4 g0;
  g0[0] = 1u;                                   // count=1, user mode
  g0[1] = lds_byte_off;                         // lds_addr (bytes)
  g0[2] = (unsigned int)ga;                     // global_addr[31:0]
  g0[3] = ((unsigned int)(ga >> 32) & 0x01FFFFFFu) | (2u << 30);  // [56:32], type=2
  i32x8 g1;
  g1[0] = (1 << 16)     // data_size = 2 bytes
        | (1 << 20)     // pad_enable
        | (3 << 22)     // pad_interval: 16 DWORDs (one 64B row)
        | (3 << 25);    // pad_amount: 4 DWORDs (16B) -> 80B row stride
  g1[1] = (tensor_w & 0xFFFF) << 16;                               // dim0[15:0]
  g1[2] = ((tensor_w >> 16) & 0xFFFF) | ((tensor_h & 0xFFFF) << 16);
  g1[3] = ((tensor_h >> 16) & 0xFFFF) | ((tile_w & 0xFFFF) << 16); // tile_dim0
  g1[4] = (tile_h & 0xFFFF);                                       // tile_dim1
  g1[5] = row_stride_elems;                                        // dim0_stride lo
  g1[6] = 0;
  g1[7] = 0;
  i32x4 z4 = {0, 0, 0, 0};
#if defined(__clang_major__) && __clang_major__ >= 23
  i32x8 z8 = {0, 0, 0, 0, 0, 0, 0, 0};
  __builtin_amdgcn_tensor_load_to_lds(g0, g1, z4, z4, z8, 0);
#else
  __builtin_amdgcn_tensor_load_to_lds(g0, g1, z4, z4, 0);
#endif
}
#else
#define HAVE_TDM 0
#endif

// ---------------------------------------------------------------------------
// Kernel 1: x (B,C,W,H) f32  ->  Xt ((B*H*W), C) bf16 via LDS tile transpose
// grid: ( (C/32)*(H/32)=32, W=64, B=16 ), block (32,8)
// ---------------------------------------------------------------------------
__global__ __launch_bounds__(256) void xpose_kernel(const float* __restrict__ x,
                                                    bf16_t* __restrict__ Xt) {
  const int gz = blockIdx.x;
  const int c0 = (gz >> 1) * 32;
  const int h0 = (gz & 1) * 32;
  const int wpos = blockIdx.y;
  const int b = blockIdx.z;
  __shared__ float tile[32][33];
  const int tx = threadIdx.x, ty = threadIdx.y;
#pragma unroll
  for (int s = 0; s < 4; ++s) {
    int c = ty + s * 8;
    tile[c][tx] = x[(((size_t)b * 512 + (c0 + c)) * 64 + wpos) * 64 + (h0 + tx)];
  }
  __syncthreads();
#pragma unroll
  for (int s = 0; s < 4; ++s) {
    int hh = ty + s * 8;
    size_t row = ((size_t)b * 64 + (h0 + hh)) * 64 + wpos;
    Xt[row * 512 + c0 + tx] = (bf16_t)tile[tx][hh];
  }
}

// ---------------------------------------------------------------------------
// Kernel 2: f32 -> bf16 elementwise (weights keep (N,K) layout: ideal B op)
// ---------------------------------------------------------------------------
__global__ __launch_bounds__(256) void cvt_bf16_kernel(const float* __restrict__ src,
                                                       bf16_t* __restrict__ dst, int n) {
  for (int i = blockIdx.x * 256 + threadIdx.x; i < n; i += gridDim.x * 256)
    dst[i] = (bf16_t)src[i];
}

// ---------------------------------------------------------------------------
// Kernel 3: emb[c][i][j] = relative[c][i - j + 63]  (c<128, i,j<64), f32
// ---------------------------------------------------------------------------
__global__ __launch_bounds__(256) void emb_kernel(const float* __restrict__ rel,
                                                  float* __restrict__ emb) {
  for (int idx = blockIdx.x * 256 + threadIdx.x; idx < 128 * 64 * 64;
       idx += gridDim.x * 256) {
    int c = idx >> 12;
    int i = (idx >> 6) & 63;
    int j = idx & 63;
    emb[idx] = rel[c * 127 + (i - j + 63)];
  }
}

// ---------------------------------------------------------------------------
// Kernel 4/6: WMMA bf16 GEMM  C[M,N] = A[M,K] * Bw[N,K]^T + bias[N]
// BM=128 BN=64 BK=32, 256 threads = 8 waves, each wave a 32x32 region.
// TDM path: double-buffered LDS, stage k+1 issued before computing stage k;
// wave 0 drains with s_wait_tensorcnt<=2 (TDM ops complete in order).
// MODE 0: store bf16 row-major (M,N).  MODE 1: scatter f32 to (B,OUT,W,H).
// ---------------------------------------------------------------------------
template <int MODE>
__global__ __launch_bounds__(256) void wmma_gemm_kernel(
    const bf16_t* __restrict__ A, const bf16_t* __restrict__ Bw,
    const float* __restrict__ bias, bf16_t* __restrict__ Cb,
    float* __restrict__ Cf, int M, int N, int K) {
  const int bm = blockIdx.y * 128;
  const int bn = blockIdx.x * 64;
  const int tid = threadIdx.x;
  const int lane = tid & 31;
  const int wv = tid >> 5;
  const int lr = lane & 15;
  const int half = lane >> 4;

  __shared__ bf16_t As[2][128][40];  // row stride 80B (16B multiple)
  __shared__ bf16_t Bs[2][64][40];

  const int m0 = (wv & 3) * 32;  // wave region within block tile
  const int n0 = (wv >> 2) * 32;

  v8f acc[2][2];
#pragma unroll
  for (int i = 0; i < 2; ++i)
#pragma unroll
    for (int j = 0; j < 2; ++j) acc[i][j] = (v8f){};

  const int nk = K >> 5;

#if HAVE_TDM
  if (wv == 0) {  // prologue: stage 0 -> buffer 0
    tdm_load_tile_bf16(A + (size_t)bm * K, (unsigned int)(uintptr_t)&As[0][0][0],
                       K, M, K, 32, 128);
    tdm_load_tile_bf16(Bw + (size_t)bn * K, (unsigned int)(uintptr_t)&Bs[0][0][0],
                       K, N, K, 32, 64);
  }
  for (int k = 0; k < nk; ++k) {
    const int cur = k & 1;
    if (wv == 0) {
      if (k + 1 < nk) {  // issue stage k+1 into the other buffer, keep in flight
        const int nxt = cur ^ 1;
        tdm_load_tile_bf16(A + (size_t)bm * K + (k + 1) * 32,
                           (unsigned int)(uintptr_t)&As[nxt][0][0], K, M, K, 32, 128);
        tdm_load_tile_bf16(Bw + (size_t)bn * K + (k + 1) * 32,
                           (unsigned int)(uintptr_t)&Bs[nxt][0][0], K, N, K, 32, 64);
        __builtin_amdgcn_s_wait_tensorcnt(2);  // stage k complete, k+1 in flight
      } else {
        __builtin_amdgcn_s_wait_tensorcnt(0);
      }
    }
    __syncthreads();
    v16bf af0 = frag_ld(&As[cur][m0 + lr][0], half);
    v16bf af1 = frag_ld(&As[cur][m0 + 16 + lr][0], half);
    v16bf bf0 = frag_ld(&Bs[cur][n0 + lr][0], half);
    v16bf bf1 = frag_ld(&Bs[cur][n0 + 16 + lr][0], half);
    acc[0][0] = wmma_bf16(af0, bf0, acc[0][0]);
    acc[0][1] = wmma_bf16(af0, bf1, acc[0][1]);
    acc[1][0] = wmma_bf16(af1, bf0, acc[1][0]);
    acc[1][1] = wmma_bf16(af1, bf1, acc[1][1]);
    __syncthreads();  // all waves done with buf[cur] before stage k+2 overwrites
  }
#else
  const int ra = tid >> 1;            // As loader row 0..127
  const int ca = (tid & 1) * 16;      // 0/16
  const int rb = tid >> 2;            // Bs loader row 0..63
  const int cb = (tid & 3) * 8;       // 0..24
  for (int k = 0; k < nk; ++k) {
    const int kb = k * 32;
    const bf16_t* arow = A + (size_t)(bm + ra) * K + kb + ca;
    *(bf16x8*)&As[0][ra][ca] = *(const bf16x8*)(arow);
    *(bf16x8*)&As[0][ra][ca + 8] = *(const bf16x8*)(arow + 8);
    const bf16_t* brow = Bw + (size_t)(bn + rb) * K + kb + cb;
    *(bf16x8*)&Bs[0][rb][cb] = *(const bf16x8*)(brow);
    if (kb + 32 < K) {  // gfx1250 global_prefetch_b8 of next K-slab
      __builtin_prefetch(arow + 32, 0, 0);
      __builtin_prefetch(brow + 32, 0, 0);
    }
    __syncthreads();
    v16bf af0 = frag_ld(&As[0][m0 + lr][0], half);
    v16bf af1 = frag_ld(&As[0][m0 + 16 + lr][0], half);
    v16bf bf0 = frag_ld(&Bs[0][n0 + lr][0], half);
    v16bf bf1 = frag_ld(&Bs[0][n0 + 16 + lr][0], half);
    acc[0][0] = wmma_bf16(af0, bf0, acc[0][0]);
    acc[0][1] = wmma_bf16(af0, bf1, acc[0][1]);
    acc[1][0] = wmma_bf16(af1, bf0, acc[1][0]);
    acc[1][1] = wmma_bf16(af1, bf1, acc[1][1]);
    __syncthreads();
  }
#endif

#pragma unroll
  for (int ti = 0; ti < 2; ++ti)
#pragma unroll
    for (int tj = 0; tj < 2; ++tj)
#pragma unroll
      for (int r = 0; r < 8; ++r) {
        int mm = bm + m0 + ti * 16 + half * 8 + r;
        int nn = bn + n0 + tj * 16 + lr;
        float v = acc[ti][tj][r] + bias[nn];
        if (MODE == 0) {
          Cb[(size_t)mm * N + nn] = (bf16_t)v;
        } else {
          // row mm = ((b*64 + h)*64 + w); out[b][nn][w][h]
          int b = mm >> 12, h = (mm >> 6) & 63, w = mm & 63;
          Cf[(((size_t)b * 512 + nn) * 64 + w) * 64 + h] = v;
        }
      }
}

// ---------------------------------------------------------------------------
// Kernel 5: fused axial attention, one block per (bh, head g). 256 thr/8 waves
// q,k: 32x64, v: 64x64, seq = 64. WMMA for QK^T and SV; VALU for qr/kr/sve.
// ---------------------------------------------------------------------------
__global__ __launch_bounds__(256) void attn_kernel(const bf16_t* __restrict__ qkv,
                                                   const float* __restrict__ emb,
                                                   bf16_t* __restrict__ attn) {
  const int blk = blockIdx.x;       // 0..8191
  const int g = blk & 7;
  const int bh = blk >> 3;
  const size_t rowbase = (size_t)bh * 64;

  const int tid = threadIdx.x;
  const int lane = tid & 31;
  const int wv = tid >> 5;
  const int lr = lane & 15;
  const int half = lane >> 4;

  __shared__ bf16_t Aq[64][40];    // q^T  (i, c)
  __shared__ bf16_t Ks[64][40];    // k^T  (j, c)
  __shared__ bf16_t Vs[64][72];    // v    (c, j)
  __shared__ float  S[64][65];     // logits / probs (i, j)
  __shared__ bf16_t SimB[64][72];  // probs bf16 (i, j)
  __shared__ float  Sve[64][65];   // sve (c, i)

  {  // stage q, k (row-major contiguous) and v (transposed into Vs)
    int r = tid >> 2;
    int c8 = (tid & 3) * 8;
    const bf16_t* qrow = qkv + (rowbase + r) * 1024 + g * 32;
    *(bf16x8*)&Aq[r][c8] = *(const bf16x8*)(qrow + c8);
    const bf16_t* krow = qkv + (rowbase + r) * 1024 + 256 + g * 32;
    *(bf16x8*)&Ks[r][c8] = *(const bf16x8*)(krow + c8);
    const bf16_t* vrow = qkv + (rowbase + r) * 1024 + 512 + g * 64;
    int cbase = (tid & 3) * 16;
#pragma unroll
    for (int t = 0; t < 16; ++t) Vs[cbase + t][r] = vrow[cbase + t];
  }
  __syncthreads();

  // tile assignment: 16 tiles (4x4), wave wv owns linear tiles 2*wv, 2*wv+1
  const int t0 = wv * 2;
  const int ti = t0 >> 2;
  const int m0 = ti * 16;
  const int n0a = (t0 & 3) * 16;
  const int n0b = n0a + 16;

  // ---- QK^T (M=64, N=64, K=32: one wmma per tile) ----
  {
    v16bf a = frag_ld(&Aq[m0 + lr][0], half);
    v16bf b0 = frag_ld(&Ks[n0a + lr][0], half);
    v16bf b1 = frag_ld(&Ks[n0b + lr][0], half);
    v8f acc0 = (v8f){}, acc1 = (v8f){};
    acc0 = wmma_bf16(a, b0, acc0);
    acc1 = wmma_bf16(a, b1, acc1);
#pragma unroll
    for (int r = 0; r < 8; ++r) {
      S[m0 + half * 8 + r][n0a + lr] = acc0[r];
      S[m0 + half * 8 + r][n0b + lr] = acc1[r];
    }
  }
  __syncthreads();

  // ---- qr + kr (VALU, relative embeddings from L2-resident table) ----
  for (int idx = tid; idx < 4096; idx += 256) {
    int i = idx >> 6, j = idx & 63;
    const float* eq = emb + i * 64 + j;                 // q_emb[c][i][j]
    const float* ek = emb + 32 * 4096 + j * 64 + i;     // k_emb[c][j][i]
    float s = 0.f;
#pragma unroll 4
    for (int c = 0; c < 32; ++c) {
      s += (float)Aq[i][c] * eq[c * 4096];
      s += (float)Ks[j][c] * ek[c * 4096];
    }
    S[i][j] += s;
  }
  __syncthreads();

  // ---- softmax over j (one thread per row) ----
  if (tid < 64) {
    int i = tid;
    float m = -1e30f;
    for (int j = 0; j < 64; ++j) m = fmaxf(m, S[i][j]);
    float sum = 0.f;
    for (int j = 0; j < 64; ++j) {
      float e = __expf(S[i][j] - m);
      S[i][j] = e;
      sum += e;
    }
    float inv = 1.0f / sum;
    for (int j = 0; j < 64; ++j) {
      float p = S[i][j] * inv;
      S[i][j] = p;
      SimB[i][j] = (bf16_t)p;
    }
  }
  __syncthreads();

  // ---- sve[c][i] = sum_j P[i][j] * v_emb[c][i][j] (VALU) ----
  for (int idx = tid; idx < 4096; idx += 256) {
    int c = idx >> 6, i = idx & 63;
    const float* ev = emb + (64 + c) * 4096 + i * 64;
    float s = 0.f;
#pragma unroll 4
    for (int j = 0; j < 64; ++j) s += S[i][j] * ev[j];
    Sve[c][i] = s;
  }

  // ---- SV (M=64, N=64, K=64: two k-steps) ----
  v8f d0 = (v8f){}, d1 = (v8f){};
#pragma unroll
  for (int kb = 0; kb < 64; kb += 32) {
    v16bf sa = frag_ld(&SimB[m0 + lr][kb], half);
    v16bf vb0 = frag_ld(&Vs[n0a + lr][kb], half);
    v16bf vb1 = frag_ld(&Vs[n0b + lr][kb], half);
    d0 = wmma_bf16(sa, vb0, d0);
    d1 = wmma_bf16(sa, vb1, d1);
  }
  __syncthreads();  // Sve complete across all waves

#pragma unroll
  for (int r = 0; r < 8; ++r) {
    int i = m0 + half * 8 + r;
    int cA = n0a + lr, cB = n0b + lr;
    size_t ro = (rowbase + i) * 512 + g * 64;
    attn[ro + cA] = (bf16_t)(d0[r] + Sve[cA][i]);
    attn[ro + cB] = (bf16_t)(d1[r] + Sve[cB][i]);
  }
}

// ---------------------------------------------------------------------------
// Host side
// ---------------------------------------------------------------------------
extern "C" void kernel_launch(void* const* d_in, const int* in_sizes, int n_in,
                              void* d_out, int out_size, void* d_ws, size_t ws_size,
                              hipStream_t stream) {
  (void)in_sizes; (void)n_in; (void)out_size; (void)ws_size;
  const float* x   = (const float*)d_in[0];  // (16,512,64,64)
  const float* ipw = (const float*)d_in[1];  // (1024,512)
  const float* ipb = (const float*)d_in[2];  // (1024,)
  const float* opw = (const float*)d_in[3];  // (512,512)
  const float* opb = (const float*)d_in[4];  // (512,)
  const float* rel = (const float*)d_in[5];  // (128,127)
  float* out = (float*)d_out;

  char* ws = (char*)d_ws;
  bf16_t* Xt   = (bf16_t*)ws;                 ws += (size_t)65536 * 512 * 2;   // 64 MB
  bf16_t* qkv  = (bf16_t*)ws;                 ws += (size_t)65536 * 1024 * 2;  // 128 MB
  bf16_t* attn = (bf16_t*)ws;                 ws += (size_t)65536 * 512 * 2;   // 64 MB
  bf16_t* Wq   = (bf16_t*)ws;                 ws += (size_t)1024 * 512 * 2;
  bf16_t* Wo   = (bf16_t*)ws;                 ws += (size_t)512 * 512 * 2;
  float*  embf = (float*)ws;                  ws += (size_t)128 * 64 * 64 * 4; // 2 MB

  // 1) layout transform + bf16
  xpose_kernel<<<dim3(32, 64, 16), dim3(32, 8), 0, stream>>>(x, Xt);
  // 2) weights -> bf16 (stay in (N,K) layout, perfect for WMMA B operand)
  cvt_bf16_kernel<<<512, 256, 0, stream>>>(ipw, Wq, 1024 * 512);
  cvt_bf16_kernel<<<256, 256, 0, stream>>>(opw, Wo, 512 * 512);
  // 3) relative embedding table
  emb_kernel<<<512, 256, 0, stream>>>(rel, embf);
  // 4) QKV projection: (65536x512) x (512x1024)
  wmma_gemm_kernel<0><<<dim3(1024 / 64, 65536 / 128), 256, 0, stream>>>(
      Xt, Wq, ipb, qkv, nullptr, 65536, 1024, 512);
  // 5) fused axial attention: 1024 bh-rows x 8 heads
  attn_kernel<<<8192, 256, 0, stream>>>(qkv, embf, attn);
  // 6) out projection + scatter to (B,OUT,W,H)
  wmma_gemm_kernel<1><<<dim3(512 / 64, 65536 / 128), 256, 0, stream>>>(
      attn, Wo, opb, nullptr, out, 65536, 512, 512);
}
